// Super_MultiMambaBlock_43181601194403
// MI455X (gfx1250) — matrix operand order repaired and checked by reading, not compiled, not run on hardware
//
#include <hip/hip_runtime.h>
#include <hip/hip_bf16.h>
#include <math.h>

typedef __bf16 bf16;
typedef __bf16 v16bf __attribute__((ext_vector_type(16)));
typedef float  v8f   __attribute__((ext_vector_type(8)));

#define BB   4
#define LL   576
#define EE   384
#define DIN  768
#define NST  16
#define RRR  24
#define RCC  96
#define ML   (BB*LL)      /* 2304 */
#define XZC  (2*DIN)      /* 1536 */
#define NC   8            /* scan chunks */
#define CH   (LL/NC)      /* 72 steps per chunk */

union FragU { v16bf v; uint4 q[2]; };

// ---- WMMA fragment loaders (CDNA5 16x16x32 bf16 layouts, wave32) ----
__device__ __forceinline__ v16bf load_a16x32(const bf16* __restrict__ A, int lda, int m0, int k0) {
    int lane = threadIdx.x & 31;
    int row  = lane & 15;
    int kh   = (lane >> 4) << 3;          // 0 or 8
    const bf16* p = A + (size_t)(m0 + row) * lda + k0 + kh;
    FragU u;
    u.q[0] = *(const uint4*)(p);          // K = kh .. kh+7
    u.q[1] = *(const uint4*)(p + 16);     // K = kh+16 .. kh+23
    return u.v;
}
__device__ __forceinline__ v16bf load_b32x16(const bf16* __restrict__ W, int ldw, int n0, int k0) {
    int lane = threadIdx.x & 31;
    int col  = lane & 15;
    int kh   = (lane >> 4) << 3;          // 0 or 8
    const bf16* p = W + (size_t)(n0 + col) * ldw + k0 + kh;
    FragU u;
    u.q[0] = *(const uint4*)(p);
    u.q[1] = *(const uint4*)(p + 16);
    return u.v;
}
__device__ __forceinline__ v8f wmma_bf16(v16bf a, v16bf b, v8f c) {
    return __builtin_amdgcn_wmma_f32_16x16x32_bf16(false, a, false, b, (short)0, c, false, false);
}

__device__ __forceinline__ int scan_pos(int dir, int t) {
    if (dir == 0) return t;
    if (dir == 1) return (LL - 1) - t;
    int u = (dir == 2) ? t : (LL - 1) - t;
    return (u % 24) * 24 + (u / 24);
}

// ---------------- prep kernels ----------------
__global__ void k_cast_bf16(const float* __restrict__ s, bf16* __restrict__ d, int n) {
    int i = blockIdx.x * 256 + threadIdx.x;
    if (i < n) d[i] = (bf16)s[i];
}
__global__ void k_prep_xw(const float* __restrict__ s, bf16* __restrict__ d) {
    int i = blockIdx.x * 256 + threadIdx.x;               // over 4*64*768
    if (i >= 4 * 64 * DIN) return;
    int c = i % DIN, r = (i / DIN) % 64, dd = i / (DIN * 64);
    d[i] = (r < 56) ? (bf16)s[((size_t)dd * 56 + r) * DIN + c] : (bf16)0.f;
}
__global__ void k_prep_dw(const float* __restrict__ s, bf16* __restrict__ d) {
    int i = blockIdx.x * 256 + threadIdx.x;               // over 4*768*32
    if (i >= 4 * DIN * 32) return;
    int k = i & 31, dc = i >> 5;
    d[i] = (k < RRR) ? (bf16)s[(size_t)dc * RRR + k] : (bf16)0.f;
}
__global__ void k_zero(float* __restrict__ p, int n) {
    int i = blockIdx.x * 256 + threadIdx.x;
    if (i < n) p[i] = 0.f;
}

// ---------------- layernorm of input + bf16 cast ----------------
__global__ void k_ln_in(const float* __restrict__ x, const float* __restrict__ g,
                        const float* __restrict__ b, bf16* __restrict__ out) {
    int m = blockIdx.x;                                   // 0..ML-1
    const float* row = x + (size_t)m * EE;
    int t = threadIdx.x;                                  // 128 threads
    float v0 = row[t], v1 = row[t + 128], v2 = row[t + 256];
    float s = v0 + v1 + v2;
    __shared__ float sm[4], sq[4];
    for (int o = 16; o; o >>= 1) s += __shfl_xor(s, o, 32);
    if ((t & 31) == 0) sm[t >> 5] = s;
    __syncthreads();
    s = sm[0] + sm[1] + sm[2] + sm[3];
    float mu = s * (1.f / EE);
    float d0 = v0 - mu, d1 = v1 - mu, d2 = v2 - mu;
    float q = d0 * d0 + d1 * d1 + d2 * d2;
    for (int o = 16; o; o >>= 1) q += __shfl_xor(q, o, 32);
    if ((t & 31) == 0) sq[t >> 5] = q;
    __syncthreads();
    q = sq[0] + sq[1] + sq[2] + sq[3];
    float rs = rsqrtf(q * (1.f / EE) + 1e-5f);
    bf16* orow = out + (size_t)m * EE;
    orow[t]       = (bf16)(d0 * rs * g[t]       + b[t]);
    orow[t + 128] = (bf16)(d1 * rs * g[t + 128] + b[t + 128]);
    orow[t + 256] = (bf16)(d2 * rs * g[t + 256] + b[t + 256]);
}

// ---------------- GEMM xz = h @ in_w.T  -> xzT[b][ch][l] ----------------
__global__ void k_gemm_xz(const bf16* __restrict__ A, const bf16* __restrict__ W,
                          float* __restrict__ xzT) {
    int wid = (blockIdx.x * blockDim.x + threadIdx.x) >> 5;
    int nt = wid % 96, mt = wid / 96;
    if (mt >= 144) return;
    int m0 = mt * 16, n0 = nt * 16;
    v8f acc = {0.f,0.f,0.f,0.f,0.f,0.f,0.f,0.f};
    for (int k0 = 0; k0 < EE; k0 += 32) {
        if (k0 + 32 < EE) {
            __builtin_prefetch(A + (size_t)(m0 + (threadIdx.x & 15)) * EE + k0 + 32, 0, 1);
            __builtin_prefetch(W + (size_t)(n0 + (threadIdx.x & 15)) * EE + k0 + 32, 0, 1);
        }
        acc = wmma_bf16(load_a16x32(A, EE, m0, k0), load_b32x16(W, EE, n0, k0), acc);
    }
    int lane = threadIdx.x & 31;
    int n  = n0 + (lane & 15);
    int mh = m0 + ((lane >> 4) << 3);
#pragma unroll
    for (int r = 0; r < 8; ++r) {
        int m = mh + r;
        int b = m / LL, l = m % LL;
        xzT[((size_t)b * XZC + n) * LL + l] = acc[r];
    }
}

// ---------------- per-direction depthwise conv + SiLU ----------------
__global__ void k_conv(const float* __restrict__ xzT, const float* __restrict__ conv_w,
                       const float* __restrict__ conv_b, float* __restrict__ xc,
                       bf16* __restrict__ xcb, int dir) {
    int bc = blockIdx.x;                                  // b*DIN + c
    int b = bc / DIN, c = bc % DIN;
    int t = threadIdx.x;                                  // 0..575
    __shared__ float sx[LL + 3];
    if (t < 3) sx[t] = 0.f;
    const float* src = xzT + ((size_t)b * XZC + c) * LL;
    sx[3 + t] = src[scan_pos(dir, t)];
    __syncthreads();
    const float* cw = conv_w + ((size_t)dir * DIN + c) * 4;
    float v = cw[0]*sx[t] + cw[1]*sx[t+1] + cw[2]*sx[t+2] + cw[3]*sx[t+3]
            + conv_b[(size_t)dir * DIN + c];
    v = v / (1.f + __expf(-v));
    xc[(size_t)bc * LL + t] = v;
    xcb[((size_t)b * LL + t) * DIN + c] = (bf16)v;
}

// ---------------- GEMM xdblT[m][56(pad 64)] = xc @ xw.T ----------------
__global__ void k_gemm_xdbl(const bf16* __restrict__ A, const bf16* __restrict__ W,
                            float* __restrict__ xdblT, bf16* __restrict__ dtb) {
    int wid = (blockIdx.x * blockDim.x + threadIdx.x) >> 5;
    int nt = wid & 3, mt = wid >> 2;
    if (mt >= 144) return;
    int m0 = mt * 16, n0 = nt * 16;
    v8f acc = {0.f,0.f,0.f,0.f,0.f,0.f,0.f,0.f};
    for (int k0 = 0; k0 < DIN; k0 += 32)
        acc = wmma_bf16(load_a16x32(A, DIN, m0, k0), load_b32x16(W, DIN, n0, k0), acc);
    int lane = threadIdx.x & 31;
    int n  = n0 + (lane & 15);
    int mh = m0 + ((lane >> 4) << 3);
#pragma unroll
    for (int r = 0; r < 8; ++r) {
        int m = mh + r;
        xdblT[(size_t)m * 64 + n] = acc[r];
        if (n0 < 32) dtb[(size_t)m * 32 + n] = (n < RRR) ? (bf16)acc[r] : (bf16)0.f;
    }
}

// ---------------- GEMM delta = softplus(dt @ dw.T + db) ----------------
__global__ void k_gemm_delta(const bf16* __restrict__ dtb, const bf16* __restrict__ W,
                             const float* __restrict__ db, float* __restrict__ delta, int dir) {
    int wid = (blockIdx.x * blockDim.x + threadIdx.x) >> 5;
    int nt = wid % 48, mt = wid / 48;
    if (mt >= 144) return;
    int m0 = mt * 16, n0 = nt * 16;
    v8f acc = {0.f,0.f,0.f,0.f,0.f,0.f,0.f,0.f};
    acc = wmma_bf16(load_a16x32(dtb, 32, m0, 0), load_b32x16(W, 32, n0, 0), acc);
    int lane = threadIdx.x & 31;
    int c  = n0 + (lane & 15);
    int mh = m0 + ((lane >> 4) << 3);
    float bias = db[(size_t)dir * DIN + c];
#pragma unroll
    for (int r = 0; r < 8; ++r) {
        int m = mh + r;
        int b = m / LL, l = m % LL;
        float v = acc[r] + bias;
        float sp = (v > 20.f) ? v : log1pf(__expf(v));
        delta[((size_t)b * DIN + c) * LL + l] = sp;
    }
}

// ---------------- chunked selective scan ----------------
// Phase A: per (chunk,b,c) local scan from h=0; emit end-state + sum(delta).
__global__ void k_scan_a(const float* __restrict__ delta, const float* __restrict__ xc,
                         const float* __restrict__ xdblT, const float* __restrict__ A_log,
                         float* __restrict__ hend, float* __restrict__ ssum, int dir) {
    int i = blockIdx.x * blockDim.x + threadIdx.x;        // (ck*BB + b)*DIN + c
    if (i >= BB * DIN * NC) return;
    int c  = i % DIN;
    int b  = (i / DIN) % BB;
    int ck = i / (BB * DIN);
    float a[NST], h[NST];
    const float* al = A_log + ((size_t)dir * DIN + c) * NST;
#pragma unroll
    for (int n = 0; n < NST; ++n) { a[n] = -__expf(al[n]); h[n] = 0.f; }
    const float* dlp = delta + ((size_t)b * DIN + c) * LL;
    const float* xp  = xc    + ((size_t)b * DIN + c) * LL;
    float S = 0.f;
    int t0 = ck * CH;
    for (int t = t0; t < t0 + CH; ++t) {
        float dl = dlp[t], xv = xp[t];
        const float* bc = xdblT + (size_t)(b * LL + t) * 64;
        float dx = dl * xv;
        S += dl;
#pragma unroll
        for (int n = 0; n < NST; ++n)
            h[n] = h[n] * __expf(dl * a[n]) + dx * bc[24 + n];
    }
    float* he = hend + (size_t)i * NST;
#pragma unroll
    for (int n = 0; n < NST; ++n) he[n] = h[n];
    ssum[i] = S;
}

// Phase B: per (b,c), 8-step prefix over chunks -> initial state per chunk.
__global__ void k_scan_b(const float* __restrict__ hend, const float* __restrict__ ssum,
                         const float* __restrict__ A_log, float* __restrict__ hinit, int dir) {
    int i = blockIdx.x * blockDim.x + threadIdx.x;        // b*DIN + c
    if (i >= BB * DIN) return;
    int c = i % DIN, b = i / DIN;
    float a[NST], H[NST];
    const float* al = A_log + ((size_t)dir * DIN + c) * NST;
#pragma unroll
    for (int n = 0; n < NST; ++n) { a[n] = -__expf(al[n]); H[n] = 0.f; }
    for (int ck = 0; ck < NC; ++ck) {
        size_t idx = ((size_t)ck * BB + b) * DIN + c;
        float* hi = hinit + idx * NST;
        const float* he = hend + idx * NST;
        float S = ssum[idx];
#pragma unroll
        for (int n = 0; n < NST; ++n) {
            hi[n] = H[n];
            H[n] = H[n] * __expf(a[n] * S) + he[n];
        }
    }
}

// Phase C: per (chunk,b,c) re-scan from true initial state, emit gated output.
__global__ void k_scan_c(const float* __restrict__ delta, const float* __restrict__ xc,
                         const float* __restrict__ xdblT, const float* __restrict__ xzT,
                         const float* __restrict__ A_log, const float* __restrict__ Dp,
                         const float* __restrict__ hinit, float* __restrict__ ydir, int dir) {
    int i = blockIdx.x * blockDim.x + threadIdx.x;        // (ck*BB + b)*DIN + c
    if (i >= BB * DIN * NC) return;
    int c  = i % DIN;
    int b  = (i / DIN) % BB;
    int ck = i / (BB * DIN);
    float a[NST], h[NST];
    const float* al = A_log + ((size_t)dir * DIN + c) * NST;
    const float* hi = hinit + (size_t)i * NST;
#pragma unroll
    for (int n = 0; n < NST; ++n) { a[n] = -__expf(al[n]); h[n] = hi[n]; }
    float dp = Dp[(size_t)dir * DIN + c];
    const float* dlp = delta + ((size_t)b * DIN + c) * LL;
    const float* xp  = xc    + ((size_t)b * DIN + c) * LL;
    const float* zb  = xzT + ((size_t)b * XZC + DIN + c) * LL;
    int t0 = ck * CH;
    for (int t = t0; t < t0 + CH; ++t) {
        float dl = dlp[t], xv = xp[t];
        const float* bc = xdblT + (size_t)(b * LL + t) * 64;
        float dx = dl * xv;
        float y = 0.f;
#pragma unroll
        for (int n = 0; n < NST; ++n) {
            h[n] = h[n] * __expf(dl * a[n]) + dx * bc[24 + n];
            y += h[n] * bc[40 + n];
        }
        int p = scan_pos(dir, t);
        float z = zb[p];
        float yv = (y + dp * xv) * (z / (1.f + __expf(-z)));
        ydir[((size_t)b * LL + p) * DIN + c] = yv;
    }
}

// ---------------- biattn: per-row LN + mean over L (atomic) ----------------
__global__ void k_attn_stats(const float* __restrict__ ydir, const float* __restrict__ lg,
                             const float* __restrict__ lb, float* __restrict__ g) {
    int m = blockIdx.x;                                   // 0..ML-1
    int b = m / LL;
    const float* row = ydir + (size_t)m * DIN;
    int t = threadIdx.x;                                  // 256 threads
    float v0 = row[t], v1 = row[t + 256], v2 = row[t + 512];
    float s = v0 + v1 + v2;
    __shared__ float sm[8], sq[8];
    for (int o = 16; o; o >>= 1) s += __shfl_xor(s, o, 32);
    if ((t & 31) == 0) sm[t >> 5] = s;
    __syncthreads();
    s = 0.f;
#pragma unroll
    for (int w = 0; w < 8; ++w) s += sm[w];
    float mu = s * (1.f / DIN);
    float d0 = v0 - mu, d1 = v1 - mu, d2 = v2 - mu;
    float q = d0 * d0 + d1 * d1 + d2 * d2;
    for (int o = 16; o; o >>= 1) q += __shfl_xor(q, o, 32);
    if ((t & 31) == 0) sq[t >> 5] = q;
    __syncthreads();
    q = 0.f;
#pragma unroll
    for (int w = 0; w < 8; ++w) q += sq[w];
    float rs = rsqrtf(q * (1.f / DIN) + 1e-5f);
    const float inv = 1.f / LL;
    atomicAdd(&g[(size_t)b * DIN + t],       (d0 * rs * lg[t]       + lb[t])       * inv);
    atomicAdd(&g[(size_t)b * DIN + t + 256], (d1 * rs * lg[t + 256] + lb[t + 256]) * inv);
    atomicAdd(&g[(size_t)b * DIN + t + 512], (d2 * rs * lg[t + 512] + lb[t + 512]) * inv);
}

// ---------------- biattn: gelu(g@gr.T+b) -> sigmoid(@cs.T+b) ----------------
__global__ void k_attn_vec(const float* __restrict__ g, const float* __restrict__ gw,
                           const float* __restrict__ gb, const float* __restrict__ cw,
                           const float* __restrict__ cb, float* __restrict__ attn) {
    int b = blockIdx.x;
    int t = threadIdx.x;                                  // 768 threads
    __shared__ float part[RCC][8];
    __shared__ float gg[RCC];
    const float* gv = g + (size_t)b * DIN;
    // stage 1: 8 threads per row r, each sums 96 elements
    {
        int r = t >> 3, pp = t & 7;
        const float* wr = gw + (size_t)r * DIN + pp * 96;
        const float* xv = gv + pp * 96;
        float s = 0.f;
#pragma unroll 8
        for (int c = 0; c < 96; ++c) s += wr[c] * xv[c];
        part[r][pp] = s;
    }
    __syncthreads();
    if (t < RCC) {
        float s = gb[t];
#pragma unroll
        for (int pp = 0; pp < 8; ++pp) s += part[t][pp];
        gg[t] = 0.5f * s * (1.f + erff(s * 0.70710678118f));
    }
    __syncthreads();
    float s = cb[t];
    const float* wr = cw + (size_t)t * RCC;
#pragma unroll 8
    for (int r = 0; r < RCC; ++r) s += wr[r] * gg[r];
    attn[(size_t)b * DIN + t] = 1.f / (1.f + __expf(-s));
}

__global__ void k_accum(const float* __restrict__ ydir, const float* __restrict__ attn,
                        float* __restrict__ acc, int first) {
    int i = blockIdx.x * 256 + threadIdx.x;
    if (i >= ML * DIN) return;
    int c = i % DIN;
    int b = i / (LL * DIN);
    float v = ydir[i] * attn[(size_t)b * DIN + c];
    acc[i] = first ? v : (acc[i] + v);
}

// ---------------- GEMM out = acc @ out_w.T + residual ----------------
__global__ void k_gemm_out(const bf16* __restrict__ A, const bf16* __restrict__ W,
                           const float* __restrict__ resid, float* __restrict__ out) {
    int wid = (blockIdx.x * blockDim.x + threadIdx.x) >> 5;
    int nt = wid % 24, mt = wid / 24;
    if (mt >= 144) return;
    int m0 = mt * 16, n0 = nt * 16;
    v8f acc = {0.f,0.f,0.f,0.f,0.f,0.f,0.f,0.f};
    for (int k0 = 0; k0 < DIN; k0 += 32) {
        if (k0 + 32 < DIN) {
            __builtin_prefetch(A + (size_t)(m0 + (threadIdx.x & 15)) * DIN + k0 + 32, 0, 1);
            __builtin_prefetch(W + (size_t)(n0 + (threadIdx.x & 15)) * DIN + k0 + 32, 0, 1);
        }
        acc = wmma_bf16(load_a16x32(A, DIN, m0, k0), load_b32x16(W, DIN, n0, k0), acc);
    }
    int lane = threadIdx.x & 31;
    int e  = n0 + (lane & 15);
    int mh = m0 + ((lane >> 4) << 3);
#pragma unroll
    for (int r = 0; r < 8; ++r) {
        int m = mh + r;
        size_t o = (size_t)m * EE + e;
        out[o] = resid[o] + acc[r];
    }
}

extern "C" void kernel_launch(void* const* d_in, const int* in_sizes, int n_in,
                              void* d_out, int out_size, void* d_ws, size_t ws_size,
                              hipStream_t stream) {
    const float* hidden   = (const float*)d_in[0];
    const float* in_w     = (const float*)d_in[1];
    const float* out_w    = (const float*)d_in[2];
    const float* norm_g   = (const float*)d_in[3];
    const float* norm_b   = (const float*)d_in[4];
    const float* A_log    = (const float*)d_in[5];
    const float* conv_w   = (const float*)d_in[6];
    const float* conv_b   = (const float*)d_in[7];
    const float* xproj_w  = (const float*)d_in[8];
    const float* dtproj_w = (const float*)d_in[9];
    const float* dtproj_b = (const float*)d_in[10];
    const float* Dp       = (const float*)d_in[11];
    const float* ln_g     = (const float*)d_in[12];
    const float* ln_b     = (const float*)d_in[13];
    const float* gr_w     = (const float*)d_in[14];
    const float* gr_b     = (const float*)d_in[15];
    const float* cs_w     = (const float*)d_in[16];
    const float* cs_b     = (const float*)d_in[17];
    float* out = (float*)d_out;
    (void)in_sizes; (void)n_in; (void)out_size; (void)ws_size;

    char* w = (char*)d_ws;
    size_t off = 0;
    auto alloc = [&](size_t bytes) -> char* {
        char* p = w + off;
        off = (off + bytes + 255) & ~(size_t)255;
        return p;
    };
    bf16*  h_bf    = (bf16*) alloc((size_t)ML * EE * 2);
    bf16*  inw_bf  = (bf16*) alloc((size_t)XZC * EE * 2);
    bf16*  outw_bf = (bf16*) alloc((size_t)EE * DIN * 2);
    bf16*  xw_bf   = (bf16*) alloc((size_t)4 * 64 * DIN * 2);
    bf16*  dw_bf   = (bf16*) alloc((size_t)4 * DIN * 32 * 2);
    float* xzT     = (float*)alloc((size_t)BB * XZC * LL * 4);
    float* xc_f    = (float*)alloc((size_t)BB * DIN * LL * 4);
    bf16*  xc_bf   = (bf16*) alloc((size_t)ML * DIN * 2);
    float* xdblT   = (float*)alloc((size_t)ML * 64 * 4);
    bf16*  dt_bf   = (bf16*) alloc((size_t)ML * 32 * 2);
    float* delta   = (float*)alloc((size_t)BB * DIN * LL * 4);
    float* ydir    = (float*)alloc((size_t)ML * DIN * 4);
    float* gbuf    = (float*)alloc((size_t)BB * DIN * 4);
    float* attnv   = (float*)alloc((size_t)BB * DIN * 4);
    float* acc     = (float*)alloc((size_t)ML * DIN * 4);
    bf16*  acc_bf  = (bf16*) alloc((size_t)ML * DIN * 2);
    float* hend    = (float*)alloc((size_t)NC * BB * DIN * NST * 4);
    float* hinit   = (float*)alloc((size_t)NC * BB * DIN * NST * 4);
    float* ssum    = (float*)alloc((size_t)NC * BB * DIN * 4);

    k_cast_bf16<<<(XZC * EE + 255) / 256, 256, 0, stream>>>(in_w, inw_bf, XZC * EE);
    k_cast_bf16<<<(EE * DIN + 255) / 256, 256, 0, stream>>>(out_w, outw_bf, EE * DIN);
    k_prep_xw<<<(4 * 64 * DIN + 255) / 256, 256, 0, stream>>>(xproj_w, xw_bf);
    k_prep_dw<<<(4 * DIN * 32 + 255) / 256, 256, 0, stream>>>(dtproj_w, dw_bf);
    k_ln_in<<<ML, 128, 0, stream>>>(hidden, norm_g, norm_b, h_bf);
    k_gemm_xz<<<1728, 256, 0, stream>>>(h_bf, inw_bf, xzT);

    for (int dir = 0; dir < 4; ++dir) {
        k_conv<<<BB * DIN, LL, 0, stream>>>(xzT, conv_w, conv_b, xc_f, xc_bf, dir);
        k_gemm_xdbl<<<72, 256, 0, stream>>>(xc_bf, xw_bf + (size_t)dir * 64 * DIN, xdblT, dt_bf);
        k_gemm_delta<<<864, 256, 0, stream>>>(dt_bf, dw_bf + (size_t)dir * DIN * 32, dtproj_b, delta, dir);
        k_scan_a<<<(BB * DIN * NC + 255) / 256, 256, 0, stream>>>(delta, xc_f, xdblT, A_log, hend, ssum, dir);
        k_scan_b<<<(BB * DIN + 255) / 256, 256, 0, stream>>>(hend, ssum, A_log, hinit, dir);
        k_scan_c<<<(BB * DIN * NC + 255) / 256, 256, 0, stream>>>(delta, xc_f, xdblT, xzT, A_log, Dp,
                                                                  hinit, ydir, dir);
        k_zero<<<(BB * DIN + 255) / 256, 256, 0, stream>>>(gbuf, BB * DIN);
        k_attn_stats<<<ML, 256, 0, stream>>>(ydir, ln_g, ln_b, gbuf);
        k_attn_vec<<<BB, DIN, 0, stream>>>(gbuf, gr_w, gr_b, cs_w, cs_b, attnv);
        k_accum<<<(ML * DIN + 255) / 256, 256, 0, stream>>>(ydir, attnv, acc, dir == 0);
    }

    k_cast_bf16<<<(ML * DIN + 255) / 256, 256, 0, stream>>>(acc, acc_bf, ML * DIN);
    k_gemm_out<<<432, 256, 0, stream>>>(acc_bf, outw_bf, hidden, out);
}
